// TGNBaseline_30932354466158
// MI455X (gfx1250) — matrix operand order repaired and checked by reading, not compiled, not run on hardware
//
#include <hip/hip_runtime.h>
#include <hip/hip_bf16.h>
#include <stdint.h>

#define N_NODES 100000
#define N_EDGES 1600000
#define IN_DIM  128
#define HID     256
#define HEADS   4
#define DH      64
#define OUT_DIM 128
#define ROWB    16
#define NBLK    (N_NODES / ROWB)   // 6250, exact

typedef __bf16 bf16;
typedef bf16  v16bf __attribute__((ext_vector_type(16)));
typedef float v8f   __attribute__((ext_vector_type(8)));

// ---- helpers -------------------------------------------------------------

__device__ inline bf16 f2bf(float f) {            // round-to-nearest-even
  unsigned u = __float_as_uint(f);
  u = u + 0x7FFFu + ((u >> 16) & 1u);
  unsigned short s = (unsigned short)(u >> 16);
  bf16 b; __builtin_memcpy(&b, &s, 2); return b;
}

// monotone float<->uint mapping so unsigned atomicMax == float max
__device__ inline unsigned fenc(float f) {
  unsigned u = __float_as_uint(f);
  return (u & 0x80000000u) ? ~u : (u | 0x80000000u);
}
__device__ inline float fdec(unsigned u) {
  return (u & 0x80000000u) ? __uint_as_float(u & 0x7FFFFFFFu)
                           : __uint_as_float(~u);
}

// ---- packing into CDNA5 WMMA fragment layouts ----------------------------
// A fragment (16x32 bf16): lane<16 -> row=lane, Ks {0..7}U{16..23};
// lane>=16 -> row=lane-16, Ks shifted by +8. dword j<4: K=2j; j>=4: K=16+2(j-4).
// B fragment (32x16 bf16): lane<16 -> col=lane, same K order as A; lane>=16 -> +8.

__global__ void k_pack_a(const float* __restrict__ nf, bf16* __restrict__ Ap) {
  int i = blockIdx.x * 256 + threadIdx.x;          // pair index
  if (i >= NBLK * 4 * 32 * 8) return;
  int j = i & 7, lane = (i >> 3) & 31, kt = (i >> 8) & 3, b = i >> 10;
  int row = b * 16 + (lane & 15);
  int k = kt * 32 + ((lane >= 16) ? 8 : 0) + ((j < 4) ? 2 * j : 16 + 2 * (j - 4));
  Ap[2 * i]     = f2bf(nf[(size_t)row * IN_DIM + k]);
  Ap[2 * i + 1] = f2bf(nf[(size_t)row * IN_DIM + k + 1]);
}

// W is [Nout, Kdim] row-major; B[k][n] = W[n][k]. Layout: [nt][kt][lane][j]
__global__ void k_pack_b(const float* __restrict__ W, bf16* __restrict__ Bp,
                         int Kdim, int KT, int total) {
  int i = blockIdx.x * 256 + threadIdx.x;          // pair index
  if (i >= total) return;
  int j = i & 7, lane = (i >> 3) & 31;
  int kt = (i >> 8) % KT;
  int nt = i / (256 * KT);
  int n = nt * 16 + (lane & 15);
  int k = kt * 32 + ((lane >= 16) ? 8 : 0) + ((j < 4) ? 2 * j : 16 + 2 * (j - 4));
  Bp[2 * i]     = f2bf(W[(size_t)n * Kdim + k]);
  Bp[2 * i + 1] = f2bf(W[(size_t)n * Kdim + k + 1]);
}

// ---- GEMM 1: feat = node_feats @ W_fc^T  ([N,128]x[128,256]) -------------
__global__ void __launch_bounds__(128) k_feat_gemm(const bf16* __restrict__ Ap,
                                                   const bf16* __restrict__ Bp,
                                                   float* __restrict__ feat) {
  const int blk = blockIdx.x;
  const int wave = threadIdx.x >> 5;
  const int lane = threadIdx.x & 31;
  v8f acc[4] = {};
#pragma unroll
  for (int kt = 0; kt < 4; ++kt) {
    v16bf a = *(const v16bf*)(Ap + (((size_t)blk * 4 + kt) * 32 + lane) * 16);
#pragma unroll
    for (int j = 0; j < 4; ++j) {
      const int nt = wave * 4 + j;
      v16bf b = *(const v16bf*)(Bp + (((size_t)nt * 4 + kt) * 32 + lane) * 16);
      acc[j] = __builtin_amdgcn_wmma_f32_16x16x32_bf16(false, a, false, b,
                                                       (short)0, acc[j], false, false);
    }
  }
  const int nlo = lane & 15, rhi = (lane >> 4) * 8;
#pragma unroll
  for (int j = 0; j < 4; ++j) {
    const int col = (wave * 4 + j) * 16 + nlo;
#pragma unroll
    for (int g = 0; g < 8; ++g)
      feat[(size_t)(blk * 16 + rhi + g) * HID + col] = acc[j][g];
  }
}

// ---- attn dots: el/er per (node, head) -----------------------------------
__global__ void k_attn_dots(const float* __restrict__ feat, const float* __restrict__ al,
                            const float* __restrict__ ar, float* __restrict__ el,
                            float* __restrict__ er) {
  int i = blockIdx.x * 256 + threadIdx.x;
  if (i >= N_NODES * HEADS) return;
  int n = i >> 2, h = i & 3;
  const float* f = feat + (size_t)n * HID + h * DH;
  const float* a = al + h * DH;
  const float* b = ar + h * DH;
  float sl = 0.f, sr = 0.f;
  for (int d = 0; d < DH; ++d) { float v = f[d]; sl += v * a[d]; sr += v * b[d]; }
  el[i] = sl; er[i] = sr;
}

__global__ void k_init_emax(unsigned* __restrict__ emax) {
  int i = blockIdx.x * 256 + threadIdx.x;
  if (i < N_NODES * HEADS) emax[i] = 0x00800000u;   // fenc(-FLT_MAX)
}

__device__ inline float edge_e(const float* el, const float* er, int s, int d, int h) {
  float x = el[s * HEADS + h] + er[d * HEADS + h];
  return (x > 0.f) ? x : 0.2f * x;                  // leaky_relu(0.2)
}

__global__ void k_edge_max(const int* __restrict__ src, const int* __restrict__ dst,
                           const float* __restrict__ el, const float* __restrict__ er,
                           unsigned* __restrict__ emax) {
  int i = blockIdx.x * 256 + threadIdx.x;
  if (i >= N_EDGES * HEADS) return;
  int e = i >> 2, h = i & 3;
  float x = edge_e(el, er, src[e], dst[e], h);
  atomicMax(&emax[dst[e] * HEADS + h], fenc(x));
}

__global__ void k_edge_denom(const int* __restrict__ src, const int* __restrict__ dst,
                             const float* __restrict__ el, const float* __restrict__ er,
                             const unsigned* __restrict__ emax, float* __restrict__ denom) {
  int i = blockIdx.x * 256 + threadIdx.x;
  if (i >= N_EDGES * HEADS) return;
  int e = i >> 2, h = i & 3;
  int d = dst[e];
  float x = edge_e(el, er, src[e], d, h);
  float m = fdec(emax[d * HEADS + h]);
  atomicAdd(&denom[d * HEADS + h], __expf(x - m));
}

// ---- edge aggregation: rst[dst] += alpha * feat[src] (wave per edge) -----
__global__ void __launch_bounds__(128) k_edge_aggr(
    const int* __restrict__ src, const int* __restrict__ dst,
    const float* __restrict__ el, const float* __restrict__ er,
    const unsigned* __restrict__ emax, const float* __restrict__ denom,
    const float* __restrict__ feat, float* __restrict__ rst) {
  int e = blockIdx.x * 4 + (threadIdx.x >> 5);
  if (e >= N_EDGES) return;
  int lane = threadIdx.x & 31;
  int s = src[e], d = dst[e];
  int h = lane >> 3;                                 // 8 lanes per head (64 cols)
  float x = el[s * HEADS + h] + er[d * HEADS + h];
  x = (x > 0.f) ? x : 0.2f * x;
  float m = fdec(emax[d * HEADS + h]);
  float alpha = __expf(x - m) / denom[d * HEADS + h];
  const float4* fp = (const float4*)(feat + (size_t)s * HID + lane * 8);
  float4 f0 = fp[0], f1 = fp[1];
  float* rp = rst + (size_t)d * HID + lane * 8;
  atomicAdd(rp + 0, alpha * f0.x); atomicAdd(rp + 1, alpha * f0.y);
  atomicAdd(rp + 2, alpha * f0.z); atomicAdd(rp + 3, alpha * f0.w);
  atomicAdd(rp + 4, alpha * f1.x); atomicAdd(rp + 5, alpha * f1.y);
  atomicAdd(rp + 6, alpha * f1.z); atomicAdd(rp + 7, alpha * f1.w);
}

// ---- fused GRU (hx=0) + out projection per 16-row block ------------------
__global__ void __launch_bounds__(128) k_gru_out(
    const float* __restrict__ rst, const float* __restrict__ bias_gat,
    const bf16* __restrict__ Wihp, const float* __restrict__ b_ih,
    const float* __restrict__ b_hh, const bf16* __restrict__ Woutp,
    const float* __restrict__ b_out, float* __restrict__ out) {
  __shared__ bf16 hbuf[16 * HID];       // h (phases 1-2) then h2 (phases 3-4)
  __shared__ float gis[16 * 768];       // gi
  const int blk = blockIdx.x;
  const int tid = threadIdx.x;
  const int wave = tid >> 5, lane = tid & 31;
  const int nlo = lane & 15;
  const int koff = (lane >= 16) ? 8 : 0;
  const int rhi = (lane >> 4) * 8;

  // phase 1: h = rst + bias_gat  -> bf16 LDS (row-major)
  for (int i = tid; i < 16 * HID; i += 128) {
    int r = i >> 8, c = i & 255;
    hbuf[i] = f2bf(rst[(size_t)(blk * 16 + r) * HID + c] + bias_gat[c]);
  }
  __syncthreads();

  // phase 2: gi = h @ W_ih^T  (48 n-tiles x 8 k-tiles; 12 n-tiles/wave)
  {
    v8f acc[12] = {};
    for (int kt = 0; kt < 8; ++kt) {
      union { unsigned u[8]; v16bf v; } af;
#pragma unroll
      for (int j = 0; j < 8; ++j) {
        int k = kt * 32 + koff + ((j < 4) ? 2 * j : 16 + 2 * (j - 4));
        af.u[j] = *(const unsigned*)&hbuf[nlo * HID + k];
      }
#pragma unroll
      for (int j = 0; j < 12; ++j) {
        int nt = wave * 12 + j;
        v16bf b = *(const v16bf*)(Wihp + (((size_t)nt * 8 + kt) * 32 + lane) * 16);
        acc[j] = __builtin_amdgcn_wmma_f32_16x16x32_bf16(false, af.v, false, b,
                                                         (short)0, acc[j], false, false);
      }
    }
#pragma unroll
    for (int j = 0; j < 12; ++j) {
      int col = (wave * 12 + j) * 16 + nlo;
#pragma unroll
      for (int g = 0; g < 8; ++g) gis[(rhi + g) * 768 + col] = acc[j][g];
    }
  }
  __syncthreads();

  // phase 3: GRU elementwise with hx = 0 => gh = b_hh (constant)
  for (int i = tid; i < 16 * HID; i += 128) {
    int r = i >> 8, c = i & 255;
    float ir  = gis[r * 768 + c]       + b_ih[c]       + b_hh[c];
    float iz  = gis[r * 768 + 256 + c] + b_ih[256 + c] + b_hh[256 + c];
    float inn = gis[r * 768 + 512 + c] + b_ih[512 + c];
    float rr = 1.f / (1.f + __expf(-ir));
    float zz = 1.f / (1.f + __expf(-iz));
    float nn = tanhf(inn + rr * b_hh[512 + c]);
    hbuf[i] = f2bf((1.f - zz) * nn);                 // h_mem = (1-z)*n
  }
  __syncthreads();

  // phase 4: out = h2 @ W_out^T + b_out (8 n-tiles x 8 k-tiles; 2/wave)
  {
    v8f acc[2] = {};
    for (int kt = 0; kt < 8; ++kt) {
      union { unsigned u[8]; v16bf v; } af;
#pragma unroll
      for (int j = 0; j < 8; ++j) {
        int k = kt * 32 + koff + ((j < 4) ? 2 * j : 16 + 2 * (j - 4));
        af.u[j] = *(const unsigned*)&hbuf[nlo * HID + k];
      }
#pragma unroll
      for (int j = 0; j < 2; ++j) {
        int nt = wave * 2 + j;
        v16bf b = *(const v16bf*)(Woutp + (((size_t)nt * 8 + kt) * 32 + lane) * 16);
        acc[j] = __builtin_amdgcn_wmma_f32_16x16x32_bf16(false, af.v, false, b,
                                                         (short)0, acc[j], false, false);
      }
    }
#pragma unroll
    for (int j = 0; j < 2; ++j) {
      int col = (wave * 2 + j) * 16 + nlo;
#pragma unroll
      for (int g = 0; g < 8; ++g)
        out[(size_t)(blk * 16 + rhi + g) * OUT_DIM + col] = acc[j][g] + b_out[col];
    }
  }
}

// ---- host ----------------------------------------------------------------
extern "C" void kernel_launch(void* const* d_in, const int* in_sizes, int n_in,
                              void* d_out, int out_size, void* d_ws, size_t ws_size,
                              hipStream_t stream) {
  const int*   src  = (const int*)d_in[0];
  const int*   dst  = (const int*)d_in[1];
  const float* nf   = (const float*)d_in[2];
  // d_in[3] edge_feats, [4] timestamps, [5] W_t, [6] b_t: dead in reference
  const float* Wfc  = (const float*)d_in[7];
  const float* al   = (const float*)d_in[8];
  const float* ar   = (const float*)d_in[9];
  const float* bgat = (const float*)d_in[10];
  const float* Wih  = (const float*)d_in[11];
  // d_in[12] W_hh: dead (hx == 0)
  const float* bih  = (const float*)d_in[13];
  const float* bhh  = (const float*)d_in[14];
  const float* Wout = (const float*)d_in[15];
  const float* bout = (const float*)d_in[16];
  float* out = (float*)d_out;

  char* ws = (char*)d_ws;
  size_t off = 0;
  auto alloc = [&](size_t bytes) -> void* {
    void* p = ws + off;
    off += bytes;
    off = (off + 255) & ~(size_t)255;
    return p;
  };

  bf16*     Ap    = (bf16*)    alloc((size_t)NBLK * 4 * 512 * sizeof(bf16));
  float*    feat  = (float*)   alloc((size_t)N_NODES * HID * sizeof(float));
  float*    rst   = (float*)   alloc((size_t)N_NODES * HID * sizeof(float));
  float*    el    = (float*)   alloc((size_t)N_NODES * HEADS * sizeof(float));
  float*    er    = (float*)   alloc((size_t)N_NODES * HEADS * sizeof(float));
  unsigned* emax  = (unsigned*)alloc((size_t)N_NODES * HEADS * sizeof(unsigned));
  float*    denom = (float*)   alloc((size_t)N_NODES * HEADS * sizeof(float));
  bf16*     Wfcp  = (bf16*)    alloc((size_t)16 * 4 * 512 * sizeof(bf16));
  bf16*     Wihp  = (bf16*)    alloc((size_t)48 * 8 * 512 * sizeof(bf16));
  bf16*     Woutp = (bf16*)    alloc((size_t)8 * 8 * 512 * sizeof(bf16));

  (void)hipMemsetAsync(rst,   0, (size_t)N_NODES * HID * sizeof(float), stream);
  (void)hipMemsetAsync(denom, 0, (size_t)N_NODES * HEADS * sizeof(float), stream);
  k_init_emax<<<(N_NODES * HEADS + 255) / 256, 256, 0, stream>>>(emax);

  k_pack_a<<<(NBLK * 1024 + 255) / 256, 256, 0, stream>>>(nf, Ap);
  k_pack_b<<<(16 * 4 * 256 + 255) / 256, 256, 0, stream>>>(Wfc,  Wfcp,  IN_DIM, 4, 16 * 4 * 256);
  k_pack_b<<<(48 * 8 * 256 + 255) / 256, 256, 0, stream>>>(Wih,  Wihp,  HID,    8, 48 * 8 * 256);
  k_pack_b<<<(8  * 8 * 256 + 255) / 256, 256, 0, stream>>>(Wout, Woutp, HID,    8, 8 * 8 * 256);

  k_feat_gemm<<<NBLK, 128, 0, stream>>>(Ap, Wfcp, feat);
  k_attn_dots<<<(N_NODES * HEADS + 255) / 256, 256, 0, stream>>>(feat, al, ar, el, er);

  k_edge_max  <<<(N_EDGES * HEADS + 255) / 256, 256, 0, stream>>>(src, dst, el, er, emax);
  k_edge_denom<<<(N_EDGES * HEADS + 255) / 256, 256, 0, stream>>>(src, dst, el, er, emax, denom);
  k_edge_aggr <<<N_EDGES / 4, 128, 0, stream>>>(src, dst, el, er, emax, denom, feat, rst);

  k_gru_out<<<NBLK, 128, 0, stream>>>(rst, bgat, Wihp, bih, bhh, Woutp, bout, out);
}
